// InputModule_21397527069130
// MI455X (gfx1250) — compile-verified, matrix operand order checked by reading, not run on hardware
//
#include <hip/hip_runtime.h>
#include <hip/hip_bf16.h>

typedef _Float16 v16h  __attribute__((ext_vector_type(16)));
typedef _Float16 half8 __attribute__((ext_vector_type(8)));
typedef float    v8f   __attribute__((ext_vector_type(8)));

#define E_   300
#define H_   256
#define G3_  768
#define B_   32
#define QL_  32
#define DL_  2048
#define S_   64
#define KP_  320            // E padded to multiple of 32 for the K loop

// ---------------- workspace layout (bytes) ----------------
#define WS_FACTS16     ((size_t)0)                         // [B*S, KP] f16
#define WS_FACTSREV16  (WS_FACTS16    + (size_t)B_*S_*KP_*2)
#define WS_QEMB16      (WS_FACTSREV16 + (size_t)B_*S_*KP_*2)   // [B*QL, KP] f16
#define WS_QWIH16      (WS_QEMB16     + (size_t)B_*QL_*KP_*2)  // [G3, KP] f16
#define WS_FWIHF16     (WS_QWIH16     + (size_t)G3_*KP_*2)
#define WS_FWIHB16     (WS_FWIHF16    + (size_t)G3_*KP_*2)
#define WS_QWHH16      (WS_FWIHB16    + (size_t)G3_*KP_*2)     // [G3, H] f16
#define WS_FWHHF16     (WS_QWHH16     + (size_t)G3_*H_*2)
#define WS_FWHHB16     (WS_FWHHF16    + (size_t)G3_*H_*2)
#define WS_XGQ         (WS_FWHHB16    + (size_t)G3_*H_*2)      // [B*QL, G3] f32
#define WS_XGF         (WS_XGQ        + (size_t)B_*QL_*G3_*4)  // [B*S, G3] f32
#define WS_XGB         (WS_XGF        + (size_t)B_*S_*G3_*4)
#define WS_OUTF        (WS_XGB        + (size_t)B_*S_*G3_*4)   // [B,S,H] f32
#define WS_OUTBREV     (WS_OUTF       + (size_t)B_*S_*H_*4)
#define WS_NF          (WS_OUTBREV    + (size_t)B_*S_*H_*4)    // [B] int

// ---------------- output layout (elements) ----------------
#define OUT_QOUT   0                       // [B,1,H] f32
#define OUT_FACT   (B_*H_)                 // [B,S,H] f32
#define OUT_NF     (B_*H_ + B_*S_*H_)      // [B] int32 (raw bits)

// ---------------- WMMA helpers (ISA 7.12.2 layouts) ----------------
static __device__ __forceinline__ v8f wmma16(v16h a, v16h b, v8f c) {
  // v_wmma_f32_16x16x32_f16  D = A(16x32 f16) x B(32x16 f16) + C(f32)
  return __builtin_amdgcn_wmma_f32_16x16x32_f16(false, a, false, b, (short)0, c, false, false);
}

// Load a 16x32 tile (A: rows) or 32x16 tile (B: cols, stored row-per-N) of f16.
// lane 0-15: row r0+lane, K = k0+0..7 and k0+16..23
// lane 16-31: row r0+lane-16, K = k0+8..15 and k0+24..31
static __device__ __forceinline__ v16h load_tile(const _Float16* __restrict__ P,
                                                 int ld, int r0, int k0, int lane) {
  int m  = r0 + (lane & 15);
  int kb = k0 + ((lane >> 4) << 3);
  union { v16h v; half8 h[2]; } u;
  u.h[0] = *(const half8*)(P + (size_t)m * ld + kb);
  u.h[1] = *(const half8*)(P + (size_t)m * ld + kb + 16);
  return u.v;
}

static __device__ __forceinline__ float sigmoidf_(float x) {
  return 1.0f / (1.0f + __expf(-x));
}

// ---------------- kernels ----------------

// f32 -> f16 with K padding (rows x kdst), zero pad beyond ksrc
__global__ void cvt_f16_pad_kernel(const float* __restrict__ src, _Float16* __restrict__ dst,
                                   int ksrc, int kdst) {
  int r = blockIdx.x;
  for (int k = threadIdx.x; k < kdst; k += blockDim.x)
    dst[(size_t)r * kdst + k] = (k < ksrc) ? (_Float16)src[(size_t)r * ksrc + k] : (_Float16)0.0f;
}

// facts[b][s][:] = mean of emb over tokens of sentence s  -> f16, K-padded
__global__ void facts_kernel(const int* __restrict__ docs, const int* __restrict__ slen,
                             const float* __restrict__ emb, _Float16* __restrict__ facts16) {
  int b = blockIdx.x / S_, s = blockIdx.x % S_;
  int start = 0;
  for (int i = 0; i < s; ++i) start += slen[b * S_ + i];
  int len = slen[b * S_ + s];
  if (start > DL_) start = DL_;
  int lim = (start + len > DL_) ? (DL_ - start) : len;
  const int* drow = docs + (size_t)b * DL_ + start;
  float inv = 1.0f / (float)(len > 0 ? len : 1);
  for (int e = threadIdx.x; e < KP_; e += blockDim.x) {
    float acc = 0.0f;
    if (e < E_) {
      for (int i = 0; i < lim; ++i) {
        // keep the HBM gather pipe ahead of the accumulate loop
        if (i + 1 < lim)
          __builtin_prefetch(emb + (size_t)drow[i + 1] * E_ + e, 0, 0);
        acc += emb[(size_t)drow[i] * E_ + e];
      }
    }
    facts16[((size_t)b * S_ + s) * KP_ + e] = (_Float16)(acc * inv);
  }
}

// q_emb f16, K-padded
__global__ void qemb_kernel(const int* __restrict__ queries, const float* __restrict__ emb,
                            _Float16* __restrict__ qemb16) {
  int row = blockIdx.x;                 // b*QL + t
  int idx = queries[row];
  for (int e = threadIdx.x; e < KP_; e += blockDim.x)
    qemb16[(size_t)row * KP_ + e] = (e < E_) ? (_Float16)emb[(size_t)idx * E_ + e] : (_Float16)0.0f;
}

// num_facts + reversed facts; also writes num_facts into d_out tail as raw int32
__global__ void nf_rev_kernel(const int* __restrict__ slen, const _Float16* __restrict__ facts16,
                              _Float16* __restrict__ rev16, int* __restrict__ nf_ws,
                              int* __restrict__ nf_out) {
  int b = blockIdx.x;
  int nf = 0;
  for (int s = 0; s < S_; ++s) nf += (slen[b * S_ + s] > 0) ? 1 : 0;
  if (threadIdx.x == 0) { nf_ws[b] = nf; nf_out[b] = nf; }
  for (int t = 0; t < S_; ++t) {
    int ridx = nf - 1 - t;
    int valid = (ridx >= 0);
    int rc = valid ? ridx : 0;
    if (rc > S_ - 1) rc = S_ - 1;
    const _Float16* src = facts16 + ((size_t)b * S_ + rc) * KP_;
    _Float16* dst = rev16 + ((size_t)b * S_ + t) * KP_;
    for (int e = threadIdx.x; e < KP_; e += blockDim.x)
      dst[e] = valid ? src[e] : (_Float16)0.0f;
  }
}

// xg = X(f16,[M,KP]) @ W(f16,[G3,KP])^T + bias  -> f32 [M,G3]
// one wave computes a 16(M) x 64(N) tile; 8 waves per block.
// Software-pipelined K loop: tiles for k+32 are in flight while wmmas for k issue.
__global__ __launch_bounds__(256) void gemm_xg_kernel(const _Float16* __restrict__ X,
                                                      const _Float16* __restrict__ W,
                                                      const float* __restrict__ bias,
                                                      float* __restrict__ out, int M) {
  const int lane = threadIdx.x & 31;
  const int wave = threadIdx.x >> 5;
  const int tilesN = G3_ / 64;                     // 12
  int tile = blockIdx.x * 8 + wave;
  int tm = tile / tilesN, tn = tile % tilesN;
  if (tm * 16 >= M) return;                        // wave-uniform

  const int m0 = tm * 16, n0 = tn * 64;
  v8f acc0 = {}, acc1 = {}, acc2 = {}, acc3 = {};

  // prologue: k = 0 tiles
  v16h aC  = load_tile(X, KP_, m0, 0, lane);
  v16h b0C = load_tile(W, KP_, n0 +  0, 0, lane);
  v16h b1C = load_tile(W, KP_, n0 + 16, 0, lane);
  v16h b2C = load_tile(W, KP_, n0 + 32, 0, lane);
  v16h b3C = load_tile(W, KP_, n0 + 48, 0, lane);

#pragma unroll
  for (int k0 = 0; k0 < KP_; k0 += 32) {
    const int kn = (k0 + 32 < KP_) ? k0 + 32 : 0;  // last iter: harmless reload
    // issue next-step loads before consuming current tiles
    v16h aN  = load_tile(X, KP_, m0, kn, lane);
    v16h b0N = load_tile(W, KP_, n0 +  0, kn, lane);
    v16h b1N = load_tile(W, KP_, n0 + 16, kn, lane);
    v16h b2N = load_tile(W, KP_, n0 + 32, kn, lane);
    v16h b3N = load_tile(W, KP_, n0 + 48, kn, lane);

    acc0 = wmma16(aC, b0C, acc0);
    acc1 = wmma16(aC, b1C, acc1);
    acc2 = wmma16(aC, b2C, acc2);
    acc3 = wmma16(aC, b3C, acc3);

    aC = aN; b0C = b0N; b1C = b1N; b2C = b2N; b3C = b3N;
  }

  // D layout: VGPR r -> row (lane<16 ? r : r+8), col = lane&15
  int rowoff = (lane >> 4) << 3;
  int colin  = lane & 15;
  v8f accs[4] = {acc0, acc1, acc2, acc3};
#pragma unroll
  for (int s = 0; s < 4; ++s) {
    int col = n0 + s * 16 + colin;
    float bv = bias[col];
#pragma unroll
    for (int r = 0; r < 8; ++r) {
      int row = m0 + rowoff + r;
      out[(size_t)row * G3_ + col] = accs[s][r] + bv;
    }
  }
}

// Persistent GRU recurrence: blockIdx.x selects {query, fact-fwd, fact-bwd}.
// h kept in LDS; per step: f16-stage h, 24 WMMAs/wave over K=256, local gate math.
// Whh B-tiles for the next k-step are prefetched into a second register buffer.
__global__ __launch_bounds__(1024) void gru_kernel(
    const float* __restrict__ xgq, const float* __restrict__ xgf, const float* __restrict__ xgb,
    const _Float16* __restrict__ whhq, const _Float16* __restrict__ whhf, const _Float16* __restrict__ whhb,
    const float* __restrict__ bhhq, const float* __restrict__ bhhf, const float* __restrict__ bhhb,
    const int* __restrict__ qlen, const int* __restrict__ nf,
    float* __restrict__ outf, float* __restrict__ outbrev, float* __restrict__ qout) {

  const int which = blockIdx.x;
  const float*    xg;  const _Float16* whh; const float* bhh;
  const int*      lens; float* outs; int T;
  if (which == 0)      { xg = xgq; whh = whhq; bhh = bhhq; lens = qlen; outs = nullptr; T = QL_; }
  else if (which == 1) { xg = xgf; whh = whhf; bhh = bhhf; lens = nf;   outs = outf;    T = S_;  }
  else                 { xg = xgb; whh = whhb; bhh = bhhb; lens = nf;   outs = outbrev; T = S_;  }

  const int H16LD = H_ + 16;                      // pad to dodge LDS bank conflicts
  __shared__ float    h32[B_ * H_];               // 32 KB
  __shared__ _Float16 h16[B_ * (H_ + 16)];        // 17 KB

  const int tid  = threadIdx.x;
  const int lane = tid & 31;
  const int wave = tid >> 5;                      // 0..31
  const int mi   = wave >> 4;                     // 0..1  -> batch rows mi*16..mi*16+15
  const int hx   = wave & 15;                     // 0..15 -> h cols hx*16..hx*16+15
  const int rowoff = (lane >> 4) << 3;
  const int colin  = lane & 15;
  const int col    = hx * 16 + colin;             // h index owned by this lane

  const float bR = bhh[col], bZ = bhh[H_ + col], bN = bhh[2 * H_ + col];
  int lenv[8];
#pragma unroll
  for (int r = 0; r < 8; ++r) lenv[r] = lens[mi * 16 + rowoff + r];

  for (int i = tid; i < B_ * H_; i += 1024) h32[i] = 0.0f;
  __syncthreads();

  for (int t = 0; t < T; ++t) {
    // stage h in f16 for WMMA A-matrix
    for (int i = tid; i < B_ * H_; i += 1024) {
      int bb = i >> 8, hh = i & (H_ - 1);
      h16[bb * H16LD + hh] = (_Float16)h32[i];
    }
    __syncthreads();

    v8f aR = {}, aZ = {}, aN = {};
    // prologue: k = 0 weight tiles
    v16h wRc = load_tile(whh, H_, 0 * H_ + hx * 16, 0, lane);
    v16h wZc = load_tile(whh, H_, 1 * H_ + hx * 16, 0, lane);
    v16h wNc = load_tile(whh, H_, 2 * H_ + hx * 16, 0, lane);
#pragma unroll
    for (int k0 = 0; k0 < H_; k0 += 32) {
      const int kn = (k0 + 32 < H_) ? k0 + 32 : 0;
      v16h wRn = load_tile(whh, H_, 0 * H_ + hx * 16, kn, lane);
      v16h wZn = load_tile(whh, H_, 1 * H_ + hx * 16, kn, lane);
      v16h wNn = load_tile(whh, H_, 2 * H_ + hx * 16, kn, lane);
      v16h a = load_tile(h16, H16LD, mi * 16, k0, lane);   // LDS, low latency
      aR = wmma16(a, wRc, aR);
      aZ = wmma16(a, wZc, aZ);
      aN = wmma16(a, wNc, aN);
      wRc = wRn; wZc = wZn; wNc = wNn;
    }

#pragma unroll
    for (int r = 0; r < 8; ++r) {
      int b    = mi * 16 + rowoff + r;
      size_t x = ((size_t)b * T + t) * G3_;
      if (t + 1 < T)                                // prefetch next step's xg row
        __builtin_prefetch(xg + x + G3_ + col, 0, 0);
      float xr = xg[x + col], xz = xg[x + H_ + col], xn = xg[x + 2 * H_ + col];
      float rg = sigmoidf_(xr + aR[r] + bR);
      float zg = sigmoidf_(xz + aZ[r] + bZ);
      float ng = tanhf(xn + rg * (aN[r] + bN));
      float hp = h32[b * H_ + col];
      float hn = (1.0f - zg) * ng + zg * hp;
      if (t >= lenv[r]) hn = hp;                  // packed-sequence freeze
      h32[b * H_ + col] = hn;
      if (outs) outs[((size_t)b * S_ + t) * H_ + col] = hn;
    }
    __syncthreads();
  }

  if (which == 0)
    for (int i = tid; i < B_ * H_; i += 1024) qout[i] = h32[i];
}

// fact_out = (out_f + reverse(out_b_rev)) * valid
__global__ void combine_kernel(const float* __restrict__ outf, const float* __restrict__ outbrev,
                               const int* __restrict__ nf_ws, float* __restrict__ dst) {
  int b = blockIdx.x / S_, t = blockIdx.x % S_;
  int nf = nf_ws[b];
  int ridx = nf - 1 - t;
  int valid = (ridx >= 0);
  int rc = valid ? ridx : 0;
  if (rc > S_ - 1) rc = S_ - 1;
  int h = threadIdx.x;                           // 256 threads = H
  float v = 0.0f;
  if (valid)
    v = outf[((size_t)b * S_ + t) * H_ + h] + outbrev[((size_t)b * S_ + rc) * H_ + h];
  dst[((size_t)b * S_ + t) * H_ + h] = v;
}

// ---------------- launcher ----------------
extern "C" void kernel_launch(void* const* d_in, const int* in_sizes, int n_in,
                              void* d_out, int out_size, void* d_ws, size_t ws_size,
                              hipStream_t stream) {
  const int*   queries  = (const int*)d_in[0];
  const int*   qlen     = (const int*)d_in[1];
  const int*   docs     = (const int*)d_in[2];
  const int*   slen     = (const int*)d_in[3];
  const float* emb      = (const float*)d_in[4];
  const float* q_Wih    = (const float*)d_in[5];
  const float* q_Whh    = (const float*)d_in[6];
  const float* q_bih    = (const float*)d_in[7];
  const float* q_bhh    = (const float*)d_in[8];
  const float* f_Wih_f  = (const float*)d_in[9];
  const float* f_Whh_f  = (const float*)d_in[10];
  const float* f_bih_f  = (const float*)d_in[11];
  const float* f_bhh_f  = (const float*)d_in[12];
  const float* f_Wih_b  = (const float*)d_in[13];
  const float* f_Whh_b  = (const float*)d_in[14];
  const float* f_bih_b  = (const float*)d_in[15];
  const float* f_bhh_b  = (const float*)d_in[16];

  char* ws = (char*)d_ws;
  _Float16* facts16   = (_Float16*)(ws + WS_FACTS16);
  _Float16* factsrev  = (_Float16*)(ws + WS_FACTSREV16);
  _Float16* qemb16    = (_Float16*)(ws + WS_QEMB16);
  _Float16* qwih16    = (_Float16*)(ws + WS_QWIH16);
  _Float16* fwihf16   = (_Float16*)(ws + WS_FWIHF16);
  _Float16* fwihb16   = (_Float16*)(ws + WS_FWIHB16);
  _Float16* qwhh16    = (_Float16*)(ws + WS_QWHH16);
  _Float16* fwhhf16   = (_Float16*)(ws + WS_FWHHF16);
  _Float16* fwhhb16   = (_Float16*)(ws + WS_FWHHB16);
  float*    xgq       = (float*)(ws + WS_XGQ);
  float*    xgf       = (float*)(ws + WS_XGF);
  float*    xgb       = (float*)(ws + WS_XGB);
  float*    outf      = (float*)(ws + WS_OUTF);
  float*    outbrev   = (float*)(ws + WS_OUTBREV);
  int*      nf_ws     = (int*)(ws + WS_NF);

  float* qout_dst = (float*)d_out + OUT_QOUT;
  float* fact_dst = (float*)d_out + OUT_FACT;
  int*   nf_out   = (int*)((float*)d_out + OUT_NF);

  // 1) weight conversion (f32 -> f16, K-pad where needed)
  cvt_f16_pad_kernel<<<G3_, 128, 0, stream>>>(q_Wih,   qwih16,  E_, KP_);
  cvt_f16_pad_kernel<<<G3_, 128, 0, stream>>>(f_Wih_f, fwihf16, E_, KP_);
  cvt_f16_pad_kernel<<<G3_, 128, 0, stream>>>(f_Wih_b, fwihb16, E_, KP_);
  cvt_f16_pad_kernel<<<G3_, 128, 0, stream>>>(q_Whh,   qwhh16,  H_, H_);
  cvt_f16_pad_kernel<<<G3_, 128, 0, stream>>>(f_Whh_f, fwhhf16, H_, H_);
  cvt_f16_pad_kernel<<<G3_, 128, 0, stream>>>(f_Whh_b, fwhhb16, H_, H_);

  // 2) gathers / sentence means
  facts_kernel<<<B_ * S_, 128, 0, stream>>>(docs, slen, emb, facts16);
  qemb_kernel <<<B_ * QL_, 128, 0, stream>>>(queries, emb, qemb16);
  nf_rev_kernel<<<B_, 256, 0, stream>>>(slen, facts16, factsrev, nf_ws, nf_out);

  // 3) input-gate GEMMs (WMMA): tiles = (M/16)*(G3/64), 8 waves/block
  gemm_xg_kernel<<<(B_ * QL_ / 16) * (G3_ / 64) / 8, 256, 0, stream>>>(qemb16,   qwih16,  q_bih,   xgq, B_ * QL_);
  gemm_xg_kernel<<<(B_ * S_  / 16) * (G3_ / 64) / 8, 256, 0, stream>>>(facts16,  fwihf16, f_bih_f, xgf, B_ * S_);
  gemm_xg_kernel<<<(B_ * S_  / 16) * (G3_ / 64) / 8, 256, 0, stream>>>(factsrev, fwihb16, f_bih_b, xgb, B_ * S_);

  // 4) three concurrent persistent GRU recurrences (WMMA on LDS-resident h)
  gru_kernel<<<3, 1024, 0, stream>>>(xgq, xgf, xgb, qwhh16, fwhhf16, fwhhb16,
                                     q_bhh, f_bhh_f, f_bhh_b, qlen, nf_ws,
                                     outf, outbrev, qout_dst);

  // 5) combine fwd + reversed bwd, mask invalid facts
  combine_kernel<<<B_ * S_, H_, 0, stream>>>(outf, outbrev, nf_ws, fact_dst);
}
